// MoM_5763846111249
// MI455X (gfx1250) — matrix operand order repaired and checked by reading, not compiled, not run on hardware
//
#include <hip/hip_runtime.h>
#include <stdint.h>

#define SEQ   4096
#define BATCH 16
#define DIM   512
#define HID   128
#define NMEM  8
#define NSLOT 9            // NMEM+1
#define NK    1152         // (NMEM+1)*HID
#define NKV   2304         // 2*NK
#define NTOT  2464         // 1152 K + 1152 V + 128 Q + 8 G + 24 pad  (= 77*32)
#define COLS  8            // state columns per workgroup
#define SROW  9            // padded LDS row stride (bank-conflict avoidance)

typedef __attribute__((ext_vector_type(16))) __bf16       v16bf;
typedef __attribute__((ext_vector_type(8)))  float        v8f;
typedef __attribute__((ext_vector_type(4)))  unsigned int v4u;

union Frag { v4u q[2]; v16bf v; };

__device__ __forceinline__ unsigned short f2bf(float f) {
  unsigned int u = __float_as_uint(f);
  u += 0x7FFFu + ((u >> 16) & 1u);         // round-to-nearest-even
  return (unsigned short)(u >> 16);
}
__device__ __forceinline__ float bf2f(unsigned short h) {
  return __uint_as_float(((unsigned int)h) << 16);
}

// ---------------- pack kernels: fp32 -> bf16 staging ----------------
__global__ void pack_x_kernel(const float* __restrict__ X,
                              unsigned short* __restrict__ Xb, int n) {
  int i = blockIdx.x * blockDim.x + threadIdx.x;
  int stride = gridDim.x * blockDim.x;
  for (; i < n; i += stride) Xb[i] = f2bf(X[i]);
}

__global__ void pack_w_kernel(const float* __restrict__ Wk, const float* __restrict__ Wv,
                              const float* __restrict__ Wq, const float* __restrict__ Wg,
                              unsigned short* __restrict__ Wall, int n) {
  int i = blockIdx.x * blockDim.x + threadIdx.x;
  if (i >= n) return;
  int row = i / DIM, k = i - row * DIM;
  float v;
  if      (row < 1152) v = Wk[row * DIM + k];
  else if (row < 2304) v = Wv[(row - 1152) * DIM + k];
  else if (row < 2432) v = Wq[(row - 2304) * DIM + k];
  else if (row < 2440) v = Wg[(row - 2432) * DIM + k];
  else                 v = 0.0f;
  Wall[i] = f2bf(v);
}

__global__ void pack_b_kernel(const float* __restrict__ bk, const float* __restrict__ bv,
                              const float* __restrict__ bq, const float* __restrict__ bg,
                              float* __restrict__ ball, int n) {
  int i = blockIdx.x * blockDim.x + threadIdx.x;
  if (i >= n) return;
  float v;
  if      (i < 1152) v = bk[i];
  else if (i < 2304) v = bv[i - 1152];
  else if (i < 2432) v = bq[i - 2304];
  else if (i < 2440) v = bg[i - 2432];
  else               v = 0.0f;
  ball[i] = v;
}

// ---------------- WMMA GEMM: Y = Xb(65536x512) @ Wall^T(512x2464) + ball ---------
// Register-blocked: each wave computes a 64x32 tile = 4(M) x 2(N) subtiles of
// 16x16, accumulated by v_wmma_f32_16x16x32_bf16 over K=512 (16 k-steps).
// Per k-step: 8 A-loads + 4 B-loads feed 8 WMMAs (1.5 b128 loads per WMMA).
// Block = 8 waves stacked along M -> 512x32 block tile; B fragments shared via L0.
__global__ void __launch_bounds__(256)
gemm_kernel(const unsigned short* __restrict__ Xb,
            const unsigned short* __restrict__ Wall,
            const float* __restrict__ ball,
            unsigned short* __restrict__ Ykv,
            float* __restrict__ Yq,
            float* __restrict__ Yg) {
  const int lane   = threadIdx.x & 31;
  const int wave   = threadIdx.x >> 5;
  const int m0     = blockIdx.x * 512 + wave * 64;   // wave's first row
  const int n0     = blockIdx.y * 32;                // wave's first col
  const int lane15 = lane & 15;
  const int half   = lane >> 4;

  // A lanes>=16 hold K=8..15,24..31 ; B lanes>=16 hold K=8..15,24..31
  const unsigned short* arow = Xb   + (size_t)(m0 + lane15) * DIM + (half ? 8 : 0);
  const unsigned short* brow = Wall + (size_t)(n0 + lane15) * DIM + (half ? 8 : 0);

  v8f c[4][2];
#pragma unroll
  for (int mi = 0; mi < 4; ++mi)
#pragma unroll
    for (int ni = 0; ni < 2; ++ni) c[mi][ni] = {};

#pragma unroll 2
  for (int k0 = 0; k0 < DIM; k0 += 32) {
    Frag a[4], b[2];
#pragma unroll
    for (int mi = 0; mi < 4; ++mi) {
      const unsigned short* p = arow + (size_t)mi * 16 * DIM + k0;
      a[mi].q[0] = *(const v4u*)(p);
      a[mi].q[1] = *(const v4u*)(p + 16);
    }
#pragma unroll
    for (int ni = 0; ni < 2; ++ni) {
      const unsigned short* p = brow + (size_t)ni * 16 * DIM + k0;
      b[ni].q[0] = *(const v4u*)(p);
      b[ni].q[1] = *(const v4u*)(p + 16);
    }
#pragma unroll
    for (int mi = 0; mi < 4; ++mi)
#pragma unroll
      for (int ni = 0; ni < 2; ++ni)
        c[mi][ni] = __builtin_amdgcn_wmma_f32_16x16x32_bf16(
            false, a[mi].v, false, b[ni].v, (short)0, c[mi][ni], false, false);
  }

  // epilogue: C layout per subtile: VGPR r -> row (mbase+r), lane -> col
#pragma unroll
  for (int ni = 0; ni < 2; ++ni) {
    const int   col  = n0 + ni * 16 + lane15;
    const float bias = ball[col];
#pragma unroll
    for (int mi = 0; mi < 4; ++mi) {
      const int mbase = m0 + mi * 16 + half * 8;
      if (col < NKV) {
#pragma unroll
        for (int r = 0; r < 8; ++r)
          Ykv[(size_t)(mbase + r) * NKV + col] = f2bf(c[mi][ni][r] + bias);
      } else if (col < 2432) {
#pragma unroll
        for (int r = 0; r < 8; ++r)
          Yq[(size_t)(mbase + r) * HID + (col - 2304)] = c[mi][ni][r] + bias;
      } else if (col < 2448) {
#pragma unroll
        for (int r = 0; r < 8; ++r)
          Yg[(size_t)(mbase + r) * 16 + (col - 2432)] = c[mi][ni][r] + bias;
      } // cols >= 2448 are pure padding: dropped
    }
  }
}

// ---------------- routing: top-2 of 8 logits; g1 = sigmoid(s1 - s2) --------------
__global__ void routing_kernel(const float* __restrict__ Yg, int4* __restrict__ rec) {
  int r = blockIdx.x * blockDim.x + threadIdx.x;
  if (r >= SEQ * BATCH) return;
  const float* s = Yg + (size_t)r * 16;
  float v[NMEM];
#pragma unroll
  for (int j = 0; j < NMEM; ++j) v[j] = s[j];
  int i1 = 0; float s1 = v[0];
#pragma unroll
  for (int j = 1; j < NMEM; ++j) if (v[j] > s1) { s1 = v[j]; i1 = j; }
  int i2 = -1; float s2 = -3.4e38f;
#pragma unroll
  for (int j = 0; j < NMEM; ++j) if (j != i1 && v[j] > s2) { s2 = v[j]; i2 = j; }
  float g1 = 1.0f / (1.0f + __expf(s2 - s1));
  float g2 = 1.0f - g1;
  int4 o;
  o.x = __float_as_int(g1); o.y = __float_as_int(g2);
  o.z = i1 + 1;             o.w = i2 + 1;
  rec[r] = o;
}

// ---------------- sequential scan: LDS-resident column-sliced state --------------
// grid = (16 column blocks, 16 batches); each WG owns M[b, 0..8, 0..127, col0..col0+7]
// in LDS (40.5 KB, row stride padded to 9 floats). Per step: rank-1 update of the 3
// routed slots, barrier, column-wise q.M readout with wave shuffle reduction.
// Two-stage register prefetch hides the HBM latency of the bf16 K/V stream.
__global__ void __launch_bounds__(256)
scan_kernel(const unsigned short* __restrict__ Ykv,
            const float* __restrict__ Yq,
            const int4* __restrict__ rec,
            const float* __restrict__ M0,
            float* __restrict__ out) {
  __shared__ float S[NSLOT * HID * SROW];
  const int tid  = threadIdx.x;
  const int col0 = blockIdx.x * COLS;
  const int b    = blockIdx.y;

  // init all 9 slots from broadcast M_0
  for (int c = tid; c < NSLOT * HID * COLS; c += 256) {
    int s = c / (HID * COLS);
    int rm = c - s * HID * COLS;
    int i = rm / COLS;
    int j = rm - i * COLS;
    S[(s * HID + i) * SROW + j] = M0[i * HID + col0 + j];
  }
  __syncthreads();

  const int ju   = tid & 7;    // update: column
  const int iu   = tid >> 3;   // update: row group base (0..31)
  const int lane = tid & 31;
  const int jr   = tid >> 5;   // readout: wave -> column (0..7)

  int4 rA = rec[b];                 // step 0
  int4 rB = rec[BATCH + b];         // step 1
  float kc[3][4], vc[3], qc[4];
  {
    const int row0 = b;
    int sl[3] = { 0, rA.z, rA.w };
#pragma unroll
    for (int s3 = 0; s3 < 3; ++s3) {
      const size_t kb = (size_t)row0 * NKV + sl[s3] * HID;
#pragma unroll
      for (int r = 0; r < 4; ++r) kc[s3][r] = bf2f(Ykv[kb + iu + 32 * r]);
      vc[s3] = bf2f(Ykv[(size_t)row0 * NKV + NK + sl[s3] * HID + col0 + ju]);
    }
#pragma unroll
    for (int r = 0; r < 4; ++r) qc[r] = Yq[(size_t)row0 * HID + lane * 4 + r];
  }

  for (int t = 0; t < SEQ; ++t) {
    const int row   = t * BATCH + b;
    const int rowN  = ((t + 1 < SEQ) ? t + 1 : SEQ - 1) * BATCH + b;
    const int rowN2 = ((t + 2 < SEQ) ? t + 2 : SEQ - 1) * BATCH + b;

    // ---- prefetch step t+2's record + step t+1's k/v/q (addresses from rB) ----
    int4 rC = rec[rowN2];
    float kn[3][4], vn[3], qn[4];
    {
      int sl[3] = { 0, rB.z, rB.w };
#pragma unroll
      for (int s3 = 0; s3 < 3; ++s3) {
        const size_t kb = (size_t)rowN * NKV + sl[s3] * HID;
#pragma unroll
        for (int r = 0; r < 4; ++r) kn[s3][r] = bf2f(Ykv[kb + iu + 32 * r]);
        vn[s3] = bf2f(Ykv[(size_t)rowN * NKV + NK + sl[s3] * HID + col0 + ju]);
      }
#pragma unroll
      for (int r = 0; r < 4; ++r) qn[r] = Yq[(size_t)rowN * HID + lane * 4 + r];
    }

    const float g1 = __int_as_float(rA.x);
    const float g2 = __int_as_float(rA.y);
    const int   i1 = rA.z, i2 = rA.w;

    // ---- update: M[s] += k_s v_s^T for s in {0, i1, i2} ----
    {
      int sl[3] = { 0, i1, i2 };
#pragma unroll
      for (int s3 = 0; s3 < 3; ++s3) {
        const float v = vc[s3];
        const int sbase = sl[s3] * HID;
#pragma unroll
        for (int r = 0; r < 4; ++r) {
          const int i = iu + 32 * r;
          S[(sbase + i) * SROW + ju] += kc[s3][r] * v;
        }
      }
    }
    __syncthreads();

    // ---- readout: o_j = sum_i q_i (M0 + g1 Mi1 + g2 Mi2)[i, j] ----
    float acc = 0.0f;
#pragma unroll
    for (int r = 0; r < 4; ++r) {
      const int i = lane * 4 + r;
      const float m = S[(i)            * SROW + jr]        // slot 0
                    + g1 * S[(i1 * HID + i) * SROW + jr]
                    + g2 * S[(i2 * HID + i) * SROW + jr];
      acc += qc[r] * m;
    }
#pragma unroll
    for (int off = 16; off > 0; off >>= 1) acc += __shfl_down(acc, off);
    if (lane == 0) out[(size_t)row * HID + col0 + jr] = acc;
    __syncthreads();

    // ---- rotate pipeline ----
    rA = rB; rB = rC;
#pragma unroll
    for (int s3 = 0; s3 < 3; ++s3) {
      vc[s3] = vn[s3];
#pragma unroll
      for (int r = 0; r < 4; ++r) kc[s3][r] = kn[s3][r];
    }
#pragma unroll
    for (int r = 0; r < 4; ++r) qc[r] = qn[r];
  }
}

// ---------------- host launcher ----------------
extern "C" void kernel_launch(void* const* d_in, const int* in_sizes, int n_in,
                              void* d_out, int out_size, void* d_ws, size_t ws_size,
                              hipStream_t stream) {
  const float* X  = (const float*)d_in[0];
  const float* M0 = (const float*)d_in[1];
  const float* Wk = (const float*)d_in[2];
  const float* bk = (const float*)d_in[3];
  const float* Wv = (const float*)d_in[4];
  const float* bv = (const float*)d_in[5];
  const float* Wg = (const float*)d_in[6];
  const float* bg = (const float*)d_in[7];
  const float* Wq = (const float*)d_in[8];
  const float* bq = (const float*)d_in[9];
  float* out = (float*)d_out;

  char* ws = (char*)d_ws;
  size_t off = 0;
  auto alloc = [&](size_t bytes) -> char* {
    char* p = ws + off;
    off += (bytes + 255) & ~(size_t)255;
    return p;
  };
  unsigned short* Xb   = (unsigned short*)alloc((size_t)SEQ * BATCH * DIM * 2);
  unsigned short* Wall = (unsigned short*)alloc((size_t)NTOT * DIM * 2);
  float*          ball = (float*)alloc((size_t)NTOT * 4);
  unsigned short* Ykv  = (unsigned short*)alloc((size_t)SEQ * BATCH * NKV * 2);
  float*          Yq   = (float*)alloc((size_t)SEQ * BATCH * HID * 4);
  float*          Yg   = (float*)alloc((size_t)SEQ * BATCH * 16 * 4);
  int4*           rec  = (int4*)alloc((size_t)SEQ * BATCH * sizeof(int4));
  if (off > ws_size) return;  // workspace too small; deterministic no-op

  const int nX = SEQ * BATCH * DIM;
  pack_x_kernel<<<8192, 256, 0, stream>>>(X, Xb, nX);
  pack_w_kernel<<<(NTOT * DIM + 255) / 256, 256, 0, stream>>>(Wk, Wv, Wq, Wg, Wall, NTOT * DIM);
  pack_b_kernel<<<(NTOT + 255) / 256, 256, 0, stream>>>(bk, bv, bq, bg, ball, NTOT);

  gemm_kernel<<<dim3((SEQ * BATCH) / 512, NTOT / 32), 256, 0, stream>>>(
      Xb, Wall, ball, Ykv, Yq, Yg);

  routing_kernel<<<(SEQ * BATCH + 255) / 256, 256, 0, stream>>>(Yg, rec);

  scan_kernel<<<dim3(HID / COLS, BATCH), 256, 0, stream>>>(Ykv, Yq, rec, M0, out);
}
